// MultiHeadAttention_65197603553815
// MI455X (gfx1250) — matrix operand order repaired and checked by reading, not compiled, not run on hardware
//
#include <hip/hip_runtime.h>

// ---------------------------------------------------------------------------
// MultiHeadAttention for MI455X (gfx1250, wave32, WMMA + TDM).
// All matmuls: v_wmma_f32_16x16x32_bf16, f32 accumulate. Intermediates bf16.
// Attention Q/K tiles fetched LDS-direct via the Tensor Data Mover.
// ---------------------------------------------------------------------------

typedef __bf16 bf16_t;
typedef __attribute__((ext_vector_type(16))) __bf16 v16bf;
typedef __attribute__((ext_vector_type(8)))  float  v8f;

constexpr int B_  = 16;
constexpr int C_  = 256;
constexpr int H_  = 64;
constexpr int W_  = 64;
constexpr int NH_ = 8;
constexpr int HD_ = 32;
constexpr int S_  = H_ * W_;                 // 4096
constexpr size_t NEL = (size_t)B_ * C_ * S_; // 16,777,216 elements / tensor

// ---------------------------------------------------------------------------
// TDM: async DMA of a 64x64 bf16 tile (row-major, stride 64) global -> LDS.
// D# built per CDNA5 ISA ch.8: group0 = {count, lds_addr, global_addr, type=2},
// group1 = {data_size=2B, tensor 64x64, tile 64x64, dim0 stride 64}.
// Issued from one wave; completion via S_WAIT_TENSORCNT + workgroup barrier.
// ---------------------------------------------------------------------------
__device__ static inline void tdm_load_tile64x64_bf16(const void* gptr, void* lptr)
{
    typedef unsigned int u32;
    typedef u32 u32x4 __attribute__((ext_vector_type(4)));
    typedef u32 u32x8 __attribute__((ext_vector_type(8)));

    const unsigned long long ga = (unsigned long long)gptr;
    const u32 la = (u32)(unsigned long long)lptr;   // low 32 bits = LDS offset

    u32x4 g0;
    g0[0] = 1u;                                        // count=1 (valid user D#)
    g0[1] = la;                                        // lds_addr
    g0[2] = (u32)ga;                                   // global_addr[31:0]
    g0[3] = (u32)((ga >> 32) & 0x1FFFFFFull)           // global_addr[56:32]
          | (2u << 30);                                // type = 2 ("image")

    u32x8 g1;
    g1[0] = (1u << 16);        // workgroup_mask=0, data_size=1 (2 bytes)
    g1[1] = (64u << 16);       // tensor_dim0[15:0] = 64   (bits 63:48)
    g1[2] = (64u << 16);       // tensor_dim0 hi = 0, tensor_dim1 = 64
    g1[3] = (64u << 16);       // tensor_dim1 hi = 0, tile_dim0 = 64
    g1[4] = 64u;               // tile_dim1 = 64, tile_dim2 = 0
    g1[5] = 64u;               // tensor_dim0_stride[31:0] = 64
    g1[6] = 0u;                // stride hi / tensor_dim1_stride lo (unused, 2D)
    g1[7] = 0u;

    asm volatile("tensor_load_to_lds %0, %1" :: "s"(g0), "s"(g1) : "memory");
}

// ---------------------------------------------------------------------------
// 1x1 conv == per-batch GEMM:  out[b,o,s] = sum_c W[o,c] * x[b,c,s] + bias[o]
// Block: 256 threads (8 waves); 64(chan) x 64(spatial) output tile.
// Activations staged TRANSPOSED in LDS so B-fragment reads are contiguous.
// ---------------------------------------------------------------------------
template <typename TIN, typename TOUT>
__global__ __launch_bounds__(256)
void conv1x1_gemm(const TIN* __restrict__ x, const float* __restrict__ Wm,
                  const float* __restrict__ bias, TOUT* __restrict__ out)
{
    const int s0   = blockIdx.x * 64;
    const int o0   = blockIdx.y * 64;
    const int b    = blockIdx.z;
    const int tid  = threadIdx.x;
    const int lane = tid & 31;
    const int wave = tid >> 5;

    __shared__ __align__(16) bf16_t As [64 * 32];  // weights  [o][k]
    __shared__ __align__(16) bf16_t BsT[64 * 32];  // act^T    [s][k]
    __shared__ float biasS[64];

    if (tid < 64) biasS[tid] = bias[o0 + tid];

    const int tile0 = wave * 2;        // two adjacent 16x16 tiles, same tm
    const int tm  = tile0 >> 2;
    const int tn0 = tile0 & 3;
    v8f acc0 = {};
    v8f acc1 = {};

    const TIN* xb = x + (size_t)b * C_ * S_;

    for (int k0 = 0; k0 < C_; k0 += 32) {
        __syncthreads();
        // ---- stage A: W[o0+r][k0+c] -> bf16, row-major (contiguous)
        {
            const int e = tid * 8;
            const int r = e >> 5, c = e & 31;
            const float* src = Wm + (size_t)(o0 + r) * C_ + k0 + c;
            float4 f0 = *(const float4*)(src);
            float4 f1 = *(const float4*)(src + 4);
            bf16_t* dst = As + r * 32 + c;
            dst[0] = (bf16_t)f0.x; dst[1] = (bf16_t)f0.y;
            dst[2] = (bf16_t)f0.z; dst[3] = (bf16_t)f0.w;
            dst[4] = (bf16_t)f1.x; dst[5] = (bf16_t)f1.y;
            dst[6] = (bf16_t)f1.z; dst[7] = (bf16_t)f1.w;
        }
        // ---- stage B transposed: x[b][k0+r][s0+c+i] -> BsT[(c+i)][r]
        {
            const int e = tid * 8;
            const int r = e >> 6, c = e & 63;        // r: k row, c: s col
            const TIN* src = xb + (size_t)(k0 + r) * S_ + s0 + c;
#pragma unroll
            for (int i = 0; i < 8; ++i)
                BsT[(c + i) * 32 + r] = (bf16_t)(float)src[i];
        }
        __syncthreads();

        // ---- A fragment (16x32): lanes 0-15 K{0-15}, 16-31 K{16-31}
        v16bf a;
        {
            const int r  = tm * 16 + (lane & 15);
            const int kb = (lane < 16) ? 0 : 16;
#pragma unroll
            for (int i = 0; i < 8; ++i) {
                a[i]     = As[r * 32 + kb + i];
                a[i + 8] = As[r * 32 + 8 + kb + i];
            }
        }
        // ---- B fragments (32x16): contiguous reads from transposed tile
        v16bf b0, b1;
        {
            const int kb = (lane < 16) ? 0 : 16;
            const int c0 = tn0 * 16 + (lane & 15);
#pragma unroll
            for (int i = 0; i < 16; ++i) {
                b0[i] = BsT[c0 * 32 + kb + i];
                b1[i] = BsT[(c0 + 16) * 32 + kb + i];
            }
        }
        acc0 = __builtin_amdgcn_wmma_f32_16x16x32_bf16(false, a, false, b0,
                                                       (short)0, acc0, false, false);
        acc1 = __builtin_amdgcn_wmma_f32_16x16x32_bf16(false, a, false, b1,
                                                       (short)0, acc1, false, false);
    }

    // ---- store: D layout -> VGPR r: M = r (lanes 0-15) / r+8 (lanes 16-31)
    TOUT* ob = out + (size_t)b * C_ * S_;
    const int nc0   = s0 + tn0 * 16 + (lane & 15);
    const int mbase = tm * 16 + ((lane < 16) ? 0 : 8);
#pragma unroll
    for (int r = 0; r < 8; ++r) {
        const int   m  = mbase + r;
        const float bv = biasS[m];
        ob[(size_t)(o0 + m) * S_ + nc0]      = (TOUT)(acc0[r] + bv);
        ob[(size_t)(o0 + m) * S_ + nc0 + 16] = (TOUT)(acc1[r] + bv);
    }
}

// ---------------------------------------------------------------------------
// Attention over one (b, head, hd-channel) 64x64 slice, fully LDS-resident.
//   S = (Q @ K^T)/8 ;  P = softmax_rows(S) ;  O = P @ V
// Q,K tiles DMA'd to LDS by the TDM (wave 0); V staged transposed by all
// threads (overlaps the DMA). attn f32 -> d_out, O bf16 -> workspace.
// ---------------------------------------------------------------------------
__global__ __launch_bounds__(256)
void attention64(const bf16_t* __restrict__ qh, const bf16_t* __restrict__ kh,
                 const bf16_t* __restrict__ vh, float* __restrict__ attn_out,
                 bf16_t* __restrict__ heads)
{
    const int d = blockIdx.x;
    const int n = blockIdx.y;
    const int b = blockIdx.z;
    const int c = n * HD_ + d;
    const size_t base = ((size_t)b * C_ + c) * S_;

    const int tid  = threadIdx.x;
    const int lane = tid & 31;
    const int wave = tid >> 5;

    __shared__ __align__(16) bf16_t qs [64 * 64];   // [h][w]
    __shared__ __align__(16) bf16_t ks [64 * 64];   // [g][w]
    __shared__ __align__(16) bf16_t vsT[64 * 64];   // [w][g]  (transposed!)
    __shared__ __align__(16) float  ps [64 * 64];   // logits -> attn

    // ---- wave 0 launches async TDM copies of Q and K tiles into LDS
    if (wave == 0) {
        tdm_load_tile64x64_bf16(qh + base, qs);
        tdm_load_tile64x64_bf16(kh + base, ks);
    }
    // ---- all threads stage V transposed (overlaps the in-flight DMA)
    {
        const int e = tid * 16;                  // 4096 elems, 16 per thread
        const int g = e >> 6, w0 = e & 63;       // row g, cols w0..w0+15
        const uint4* src = (const uint4*)(vh + base + (size_t)g * 64 + w0);
        uint4 p0 = src[0], p1 = src[1];
        const bf16_t* pv = (const bf16_t*)&p0;
#pragma unroll
        for (int i = 0; i < 8; ++i) vsT[(w0 + i) * 64 + g] = pv[i];
        pv = (const bf16_t*)&p1;
#pragma unroll
        for (int i = 0; i < 8; ++i) vsT[(w0 + 8 + i) * 64 + g] = pv[i];
    }
    if (wave == 0) __builtin_amdgcn_s_wait_tensorcnt(0);
    __syncthreads();

    const int tile0 = wave * 2;
    const int tm  = tile0 >> 2;
    const int tn0 = tile0 & 3;

    // ---- logits: S[h,g] = sum_w q[h,w]*k[g,w]  (B[k][n] = ks[n*64+k], contig)
    v8f acc0 = {}, acc1 = {};
#pragma unroll
    for (int kc = 0; kc < 64; kc += 32) {
        v16bf a;
        {
            const int r  = tm * 16 + (lane & 15);
            const int kb = (lane < 16) ? 0 : 16;
#pragma unroll
            for (int i = 0; i < 8; ++i) {
                a[i]     = qs[r * 64 + kc + kb + i];
                a[i + 8] = qs[r * 64 + kc + 8 + kb + i];
            }
        }
        v16bf b0, b1;
        {
            const int kb = kc + ((lane < 16) ? 0 : 16);
            const int c0 = tn0 * 16 + (lane & 15);
#pragma unroll
            for (int i = 0; i < 16; ++i) {
                b0[i] = ks[c0 * 64 + kb + i];
                b1[i] = ks[(c0 + 16) * 64 + kb + i];
            }
        }
        acc0 = __builtin_amdgcn_wmma_f32_16x16x32_bf16(false, a, false, b0,
                                                       (short)0, acc0, false, false);
        acc1 = __builtin_amdgcn_wmma_f32_16x16x32_bf16(false, a, false, b1,
                                                       (short)0, acc1, false, false);
    }
    // scaled logits -> LDS
    {
        const float scale = 0.125f;   // 1/sqrt(64)
        const int nc0 = tn0 * 16 + (lane & 15);
        const int mb  = tm * 16 + ((lane < 16) ? 0 : 8);
#pragma unroll
        for (int r = 0; r < 8; ++r) {
            ps[(mb + r) * 64 + nc0]      = acc0[r] * scale;
            ps[(mb + r) * 64 + nc0 + 16] = acc1[r] * scale;
        }
    }
    __syncthreads();

    // ---- softmax: 4 lanes per row, contiguous 16-elem chunks, shfl reduce
    {
        const int row = tid >> 2;
        const int qq  = tid & 3;
        float* rp = ps + row * 64 + qq * 16;
        float m = rp[0];
#pragma unroll
        for (int j = 1; j < 16; ++j) m = fmaxf(m, rp[j]);
        m = fmaxf(m, __shfl_xor(m, 1, 32));
        m = fmaxf(m, __shfl_xor(m, 2, 32));
        float ssum = 0.f;
#pragma unroll
        for (int j = 0; j < 16; ++j) { float e = __expf(rp[j] - m); rp[j] = e; ssum += e; }
        ssum += __shfl_xor(ssum, 1, 32);
        ssum += __shfl_xor(ssum, 2, 32);
        const float inv = 1.f / ssum;
#pragma unroll
        for (int j = 0; j < 16; ++j) rp[j] *= inv;
    }
    __syncthreads();

    // ---- stream attn (f32) straight from LDS: [b,n,d,h,g] == [b, c, :, :]
    {
        float* ao = attn_out + base;      // same linear index as (b*C + c)*4096
        const float4* src = (const float4*)ps;
        float4* dst = (float4*)ao;
#pragma unroll
        for (int i = 0; i < 4; ++i) dst[tid + 256 * i] = src[tid + 256 * i];
    }

    // ---- O = P @ V : P rows f32 -> bf16 on the fly; B-frags contiguous (vsT)
    v8f o0 = {}, o1 = {};
#pragma unroll
    for (int kc = 0; kc < 64; kc += 32) {
        v16bf a;
        {
            const int r  = tm * 16 + (lane & 15);
            const int kb = (lane < 16) ? 0 : 16;
#pragma unroll
            for (int i = 0; i < 8; ++i) {
                a[i]     = (bf16_t)ps[r * 64 + kc + kb + i];
                a[i + 8] = (bf16_t)ps[r * 64 + kc + 8 + kb + i];
            }
        }
        v16bf b0, b1;
        {
            const int kb = kc + ((lane < 16) ? 0 : 16);
            const int c0 = tn0 * 16 + (lane & 15);
#pragma unroll
            for (int i = 0; i < 16; ++i) {
                b0[i] = vsT[c0 * 64 + kb + i];
                b1[i] = vsT[(c0 + 16) * 64 + kb + i];
            }
        }
        o0 = __builtin_amdgcn_wmma_f32_16x16x32_bf16(false, a, false, b0,
                                                     (short)0, o0, false, false);
        o1 = __builtin_amdgcn_wmma_f32_16x16x32_bf16(false, a, false, b1,
                                                     (short)0, o1, false, false);
    }
    // store head output bf16
    {
        bf16_t* hb = heads + base;
        const int nc0 = tn0 * 16 + (lane & 15);
        const int mb  = tm * 16 + ((lane < 16) ? 0 : 8);
#pragma unroll
        for (int r = 0; r < 8; ++r) {
            hb[(size_t)(mb + r) * 64 + nc0]      = (bf16_t)o0[r];
            hb[(size_t)(mb + r) * 64 + nc0 + 16] = (bf16_t)o1[r];
        }
    }
}

// ---------------------------------------------------------------------------
extern "C" void kernel_launch(void* const* d_in, const int* in_sizes, int n_in,
                              void* d_out, int out_size, void* d_ws, size_t ws_size,
                              hipStream_t stream)
{
    const float* q  = (const float*)d_in[0];
    const float* k  = (const float*)d_in[1];
    const float* v  = (const float*)d_in[2];
    const float* Wq = (const float*)d_in[3];  const float* bq = (const float*)d_in[4];
    const float* Wk = (const float*)d_in[5];  const float* bk = (const float*)d_in[6];
    const float* Wv = (const float*)d_in[7];  const float* bv = (const float*)d_in[8];
    const float* Wo = (const float*)d_in[9];  const float* bo = (const float*)d_in[10];

    float* out  = (float*)d_out;          // [B,C,H,W] f32
    float* attn = out + NEL;              // [B,NH,HD,H,H] f32

    bf16_t* qh    = (bf16_t*)d_ws;        // 4 bf16 tensors = 128 MB workspace
    bf16_t* kh    = qh + NEL;
    bf16_t* vh    = kh + NEL;
    bf16_t* heads = vh + NEL;

    dim3 gproj(S_ / 64, C_ / 64, B_);     // (64, 4, 16)
    dim3 blk(256);

    conv1x1_gemm<float, bf16_t><<<gproj, blk, 0, stream>>>(q, Wq, bq, qh);
    conv1x1_gemm<float, bf16_t><<<gproj, blk, 0, stream>>>(k, Wk, bk, kh);
    conv1x1_gemm<float, bf16_t><<<gproj, blk, 0, stream>>>(v, Wv, bv, vh);

    dim3 gattn(HD_, NH_, B_);             // (32, 8, 16) = 4096 slices
    attention64<<<gattn, blk, 0, stream>>>(qh, kh, vh, attn, heads);

    conv1x1_gemm<bf16_t, float><<<gproj, blk, 0, stream>>>(heads, Wo, bo, out);
}
